// GIN_Layer_Sparse_72688026518106
// MI455X (gfx1250) — matrix operand order repaired and checked by reading, not compiled, not run on hardware
//
#include <hip/hip_runtime.h>
#include <hip/hip_bf16.h>

// ---------------------------------------------------------------------------
// GIN layer (sparse aggregation) for MI455X / gfx1250, wave32, WMMA bf16.
// ---------------------------------------------------------------------------

typedef __attribute__((ext_vector_type(16))) __bf16 v16bf;
typedef __attribute__((ext_vector_type(8)))  float  v8f;

constexpr int   Cc     = 128;   // channels (fixed by reference)
constexpr int   Gc     = 64;    // graphs   (fixed by reference)
constexpr int   WS     = 136;   // sW row stride (ushorts): 272 B = 17*16
constexpr float LN_EPS = 1e-5f;

__device__ __forceinline__ unsigned short f32_to_bf16(float f) {
    unsigned int u = __float_as_uint(f);
    unsigned int r = u + 0x7FFFu + ((u >> 16) & 1u);   // round-to-nearest-even
    return (unsigned short)(r >> 16);
}

union Frag16 {            // 16 bf16 values <-> two 16-byte LDS loads
    uint4 q[2];
    v16bf v;
};

__device__ __forceinline__ void atomic_fadd(float* p, float v) {
    __hip_atomic_fetch_add(p, v, __ATOMIC_RELAXED, __HIP_MEMORY_SCOPE_AGENT);
}

// ---------------------------------------------------------------------------
// K0: h0 = (1 + eps) * node
// ---------------------------------------------------------------------------
__global__ void k_init_h0(const float* __restrict__ node,
                          const float* __restrict__ eps,
                          float* __restrict__ h0, int total) {
    const float s = 1.0f + eps[0];
    for (int i = blockIdx.x * blockDim.x + threadIdx.x; i < total;
         i += gridDim.x * blockDim.x)
        h0[i] = s * node[i];
}

__global__ void k_zero(float* __restrict__ p, int n) {
    int i = blockIdx.x * blockDim.x + threadIdx.x;
    if (i < n) p[i] = 0.0f;
}

// ---------------------------------------------------------------------------
// K1: scatter-add aggregation. One thread per (edge, 4-channel quad):
//     1 global_load_b128 + 4 global_atomic_add_f32. node + h0 (51 MB) are
//     L2-resident (192 MB), so atomics stay on-chip.
// ---------------------------------------------------------------------------
__global__ void k_scatter(const float* __restrict__ node,
                          const int* __restrict__ ei,
                          float* __restrict__ h0, int E) {
    const int total = E * 32;            // 32 quads of 4 channels per edge
    for (int idx = blockIdx.x * blockDim.x + threadIdx.x; idx < total;
         idx += gridDim.x * blockDim.x) {
        const int e = idx >> 5;
        const int q = (idx & 31) << 2;
        const int s = ei[e];
        const int d = ei[E + e];
        const float4 v = *(const float4*)(node + (size_t)s * Cc + q);
        float* p = h0 + (size_t)d * Cc + q;
        atomic_fadd(p + 0, v.x);
        atomic_fadd(p + 1, v.y);
        atomic_fadd(p + 2, v.z);
        atomic_fadd(p + 3, v.w);
    }
}

// ---------------------------------------------------------------------------
// K2/K3: fused  out = relu( LN( A @ W^T + b ) )   (FINAL=false)
//        or     hpre = A @ W^T + b + skip, + per-graph sum/count (FINAL=true)
// Block: 128 threads (4 waves), 16 node rows, full 128 output channels.
// ---------------------------------------------------------------------------
template <bool FINAL>
__global__ __launch_bounds__(128)
void k_gemm(const float* __restrict__ A,    const float* __restrict__ W,
            const float* __restrict__ bias, const float* __restrict__ lnw,
            const float* __restrict__ lnb,  float* __restrict__ out,
            const float* __restrict__ skip, const int* __restrict__ batch,
            float* __restrict__ sum_g, float* __restrict__ cnt_g, int N) {
    __shared__ __align__(16) unsigned short sW[128 * WS]; // W[o][k] bf16
    __shared__ __align__(16) unsigned short sA[16 * 128]; // A tile bf16
    __shared__ __align__(16) float          sC[16 * 128]; // f32 result tile

    const int tid = threadIdx.x;

    // Stage W into LDS as bf16: sW[o*WS+k] = W[o*128+k]  (k = tid)
    for (int o = 0; o < 128; ++o)
        sW[o * WS + tid] = f32_to_bf16(W[o * 128 + tid]);

    const int rowbase = blockIdx.x * 16;
    for (int r = 0; r < 16; ++r) {
        const int nrow = rowbase + r;
        sA[r * 128 + tid] =
            (nrow < N) ? f32_to_bf16(A[(size_t)nrow * 128 + tid])
                       : (unsigned short)0;
    }
    __syncthreads();

    const int wave = tid >> 5, lane = tid & 31;
    const int m = lane & 15, hihalf = lane >> 4;

    // Each wave: two 16x16 output tiles, K=128 in 4 WMMA steps of K=32.
    for (int t = 0; t < 2; ++t) {
        const int n0 = wave * 32 + t * 16;
        v8f acc = {0.f, 0.f, 0.f, 0.f, 0.f, 0.f, 0.f, 0.f};
#pragma unroll
        for (int kk = 0; kk < 4; ++kk) {
            // A fragment (ISA 16-bit A layout): two aligned 16-byte chunks.
            // elements 0..7  <- K = kk*32 + hihalf*8 + (0..7)
            // elements 8..15 <- K = kk*32 + hihalf*8 + (16..23)
            Frag16 a;
            const unsigned short* ap = sA + m * 128 + kk * 32 + hihalf * 8;
            a.q[0] = *(const uint4*)(ap);
            a.q[1] = *(const uint4*)(ap + 16);

            // B fragment (N = n0 + lane%16, same K order as A): two aligned 16-byte chunks.
            Frag16 b;
            const unsigned short* bp = sW + (n0 + m) * WS + kk * 32 + hihalf * 8;
            b.q[0] = *(const uint4*)(bp);
            b.q[1] = *(const uint4*)(bp + 16);

            acc = __builtin_amdgcn_wmma_f32_16x16x32_bf16(
                false, a.v, false, b.v, (short)0, acc, false, false);
        }
        // D layout: VGPR r -> M = r + 8*(lane>=16), N = lane%16
#pragma unroll
        for (int r = 0; r < 8; ++r) {
            const int mm = r + 8 * hihalf, nn = n0 + m;
            sC[mm * 128 + nn] = acc[r] + bias[nn];
        }
    }
    __syncthreads();

    // Epilogue: each wave handles rows wave, wave+4, wave+8, wave+12.
    for (int rr = 0; rr < 4; ++rr) {
        const int row  = wave + rr * 4;
        const int nrow = rowbase + row;
        float x0 = sC[row * 128 + lane];
        float x1 = sC[row * 128 + lane + 32];
        float x2 = sC[row * 128 + lane + 64];
        float x3 = sC[row * 128 + lane + 96];

        if (!FINAL) {
            float s = x0 + x1 + x2 + x3;
            float q = x0 * x0 + x1 * x1 + x2 * x2 + x3 * x3;
#pragma unroll
            for (int o = 16; o > 0; o >>= 1) {
                s += __shfl_xor(s, o, 32);
                q += __shfl_xor(q, o, 32);
            }
            const float mean = s * (1.0f / 128.0f);
            const float var  = q * (1.0f / 128.0f) - mean * mean;
            const float rstd = rsqrtf(var + LN_EPS);
            if (nrow < N) {
                float* op = out + (size_t)nrow * 128;
                const float xs[4] = {x0, x1, x2, x3};
#pragma unroll
                for (int j = 0; j < 4; ++j) {
                    const int c = lane + 32 * j;
                    float y = (xs[j] - mean) * rstd * lnw[c] + lnb[c];
                    op[c] = fmaxf(y, 0.0f);
                }
            }
        } else {
            if (nrow < N) {
                const float* sp = skip + (size_t)nrow * 128;
                float y0 = x0 + sp[lane];
                float y1 = x1 + sp[lane + 32];
                float y2 = x2 + sp[lane + 64];
                float y3 = x3 + sp[lane + 96];
                float s  = y0 + y1 + y2 + y3;
#pragma unroll
                for (int o = 16; o > 0; o >>= 1) s += __shfl_xor(s, o, 32);
                float* op = out + (size_t)nrow * 128;
                op[lane]      = y0;
                op[lane + 32] = y1;
                op[lane + 64] = y2;
                op[lane + 96] = y3;
                if (lane == 0) {
                    const int g = batch[nrow];
                    atomic_fadd(&sum_g[g], s);
                    atomic_fadd(&cnt_g[g], 128.0f);   // count in elements
                }
            }
        }
    }
}

// ---------------------------------------------------------------------------
// Graph-norm helper kernels
// ---------------------------------------------------------------------------
__global__ void k_mean(const float* __restrict__ sum_g,
                       const float* __restrict__ cnt_g,
                       float* __restrict__ mean_g) {
    const int i = threadIdx.x;
    if (i < Gc) mean_g[i] = sum_g[i] / fmaxf(cnt_g[i], 1.0f);
}

__global__ __launch_bounds__(128)
void k_ssq(const float* __restrict__ hpre, const int* __restrict__ batch,
           const float* __restrict__ mean_g, float* __restrict__ ssq_g,
           int N) {
    const int n = blockIdx.x;
    if (n >= N) return;
    const int   g  = batch[n];
    const float mu = mean_g[g];
    const float d  = hpre[(size_t)n * 128 + threadIdx.x] - mu;
    float q = d * d;
#pragma unroll
    for (int o = 16; o > 0; o >>= 1) q += __shfl_xor(q, o, 32);
    __shared__ float part[4];
    const int wave = threadIdx.x >> 5, lane = threadIdx.x & 31;
    if (lane == 0) part[wave] = q;
    __syncthreads();
    if (threadIdx.x == 0)
        atomic_fadd(&ssq_g[g], part[0] + part[1] + part[2] + part[3]);
}

__global__ void k_rstd(const float* __restrict__ ssq_g,
                       const float* __restrict__ cnt_g,
                       float* __restrict__ rstd_g) {
    const int i = threadIdx.x;
    if (i < Gc) rstd_g[i] = rsqrtf(ssq_g[i] / fmaxf(cnt_g[i], 1.0f) + LN_EPS);
}

__global__ void k_out(const float* __restrict__ hpre,
                      const int* __restrict__ batch,
                      const float* __restrict__ mean_g,
                      const float* __restrict__ rstd_g,
                      const float* __restrict__ gnw,
                      const float* __restrict__ gnb,
                      float* __restrict__ out, int total) {
    for (int i = blockIdx.x * blockDim.x + threadIdx.x; i < total;
         i += gridDim.x * blockDim.x) {
        const int n = i >> 7, c = i & 127;
        const int g = batch[n];
        float y = (hpre[i] - mean_g[g]) * rstd_g[g] * gnw[c] + gnb[c];
        out[i] = fmaxf(y, 0.0f);
    }
}

// ---------------------------------------------------------------------------
// Launcher
// ---------------------------------------------------------------------------
extern "C" void kernel_launch(void* const* d_in, const int* in_sizes, int n_in,
                              void* d_out, int out_size, void* d_ws,
                              size_t ws_size, hipStream_t stream) {
    const float* node  = (const float*)d_in[0];
    const float* W1    = (const float*)d_in[1];
    const float* b1    = (const float*)d_in[2];
    const float* ln1w  = (const float*)d_in[3];
    const float* ln1b  = (const float*)d_in[4];
    const float* W2    = (const float*)d_in[5];
    const float* b2    = (const float*)d_in[6];
    const float* ln2w  = (const float*)d_in[7];
    const float* ln2b  = (const float*)d_in[8];
    const float* W3    = (const float*)d_in[9];
    const float* b3    = (const float*)d_in[10];
    const float* gnw   = (const float*)d_in[11];
    const float* gnb   = (const float*)d_in[12];
    const float* eps   = (const float*)d_in[13];
    const int*   ei    = (const int*)d_in[14];
    const int*   batch = (const int*)d_in[15];

    const int N = in_sizes[0] / Cc;
    const int E = in_sizes[14] / 2;
    const int total = N * Cc;

    float* bufA  = (float*)d_ws;                 // h0 / h2
    float* bufB  = bufA + (size_t)N * Cc;        // h1 / hpre
    float* stats = bufB + (size_t)N * Cc;
    float* sum_g  = stats;
    float* cnt_g  = stats + Gc;
    float* mean_g = stats + 2 * Gc;
    float* ssq_g  = stats + 3 * Gc;
    float* rstd_g = stats + 4 * Gc;

    const int tgrid    = (total + 255) / 256;
    const int gemmGrid = (N + 15) / 16;
    const int scatGrid = (E * 32 + 255) / 256;

    k_zero<<<1, 512, 0, stream>>>(stats, 5 * Gc);
    k_init_h0<<<tgrid, 256, 0, stream>>>(node, eps, bufA, total);
    k_scatter<<<scatGrid, 256, 0, stream>>>(node, ei, bufA, E);

    // GEMM1: bufA -> bufB (LN+ReLU)
    k_gemm<false><<<gemmGrid, 128, 0, stream>>>(bufA, W1, b1, ln1w, ln1b,
                                                bufB, nullptr, nullptr,
                                                nullptr, nullptr, N);
    // GEMM2: bufB -> bufA (LN+ReLU)
    k_gemm<false><<<gemmGrid, 128, 0, stream>>>(bufB, W2, b2, ln2w, ln2b,
                                                bufA, nullptr, nullptr,
                                                nullptr, nullptr, N);
    // GEMM3: bufA -> bufB (= node + h3), per-graph sum/count
    k_gemm<true><<<gemmGrid, 128, 0, stream>>>(bufA, W3, b3, nullptr, nullptr,
                                               bufB, node, batch, sum_g,
                                               cnt_g, N);

    k_mean<<<1, 64, 0, stream>>>(sum_g, cnt_g, mean_g);
    k_ssq<<<N, 128, 0, stream>>>(bufB, batch, mean_g, ssq_g, N);
    k_rstd<<<1, 64, 0, stream>>>(ssq_g, cnt_g, rstd_g);
    k_out<<<tgrid, 256, 0, stream>>>(bufB, batch, mean_g, rstd_g, gnw, gnb,
                                     (float*)d_out, total);
}
